// DeformableTransformer_20048907337746
// MI455X (gfx1250) — matrix operand order repaired and checked by reading, not compiled, not run on hardware
//
#include <hip/hip_runtime.h>
#include <hip/hip_bf16.h>
#include <stdint.h>

// ---------------- problem constants ----------------
constexpr int cB   = 2;
constexpr int cD   = 256;
constexpr int cNH  = 8;
constexpr int cDH  = 32;
constexpr int cNP  = 4;
constexpr int cLV  = 4;
constexpr int cNQ  = 4096;
constexpr int cNL  = 6;
constexpr int cDFF = 1024;
constexpr int cS   = 21760;          // 128^2 + 64^2 + 32^2 + 16^2

typedef __attribute__((ext_vector_type(16))) __bf16 v16bf;
typedef __attribute__((ext_vector_type(8)))  __bf16 v8bf;
typedef __attribute__((ext_vector_type(8)))  float  v8f;

__device__ __forceinline__ unsigned short f2bf(float f) {
  unsigned int u = __float_as_uint(f);
  u += 0x7FFFu + ((u >> 16) & 1u);          // round-to-nearest-even
  return (unsigned short)(u >> 16);
}

__device__ __forceinline__ v16bf mk16(v8bf lo, v8bf hi) {
  v16bf r;
#pragma unroll
  for (int i = 0; i < 8; ++i) { r[i] = lo[i]; r[i + 8] = hi[i]; }
  return r;
}

// ---------------- fp32 -> bf16 convert ----------------
__global__ __launch_bounds__(256) void k_convert_bf16(const float* __restrict__ src,
                                                      unsigned short* __restrict__ dst,
                                                      long n) {
  long i = (long)blockIdx.x * blockDim.x + threadIdx.x;
  if (i < n) dst[i] = f2bf(src[i]);
}

// ---------------- (qf + q_pos) -> bf16 ----------------
__global__ __launch_bounds__(256) void k_add_bf16(const float* __restrict__ a,
                                                  const float* __restrict__ b,
                                                  unsigned short* __restrict__ dst,
                                                  long n) {
  long i = (long)blockIdx.x * blockDim.x + threadIdx.x;
  if (i < n) dst[i] = f2bf(a[i] + b[i]);
}

// -------- flatten (B,D,H,W) levels -> (B,S,D) bf16, tiled transpose --------
__global__ __launch_bounds__(256) void k_flatten(const float* __restrict__ s0,
                                                 const float* __restrict__ s1,
                                                 const float* __restrict__ s2,
                                                 const float* __restrict__ s3,
                                                 unsigned short* __restrict__ dst) {
  __shared__ float tile[32][33];
  const int pix0 = blockIdx.x * 32;          // S/32 = 680 tiles, level-aligned
  const int d0   = blockIdx.y * 32;          // 8 channel tiles
  const int b    = blockIdx.z;

  const float* src; int hw; int lofs;
  if      (pix0 < 16384) { src = s0; hw = 16384; lofs = 0;     }
  else if (pix0 < 20480) { src = s1; hw = 4096;  lofs = 16384; }
  else if (pix0 < 21504) { src = s2; hw = 1024;  lofs = 20480; }
  else                   { src = s3; hw = 256;   lofs = 21504; }
  const int p0 = pix0 - lofs;
  const int tx = threadIdx.x;                // 0..31
  const int ty = threadIdx.y;                // 0..7
#pragma unroll
  for (int i = 0; i < 4; ++i) {
    int dch = d0 + ty + i * 8;
    tile[ty + i * 8][tx] = src[((long)b * cD + dch) * hw + p0 + tx];
  }
  __syncthreads();
#pragma unroll
  for (int i = 0; i < 4; ++i) {
    int prow = ty + i * 8;
    dst[((long)b * cS + pix0 + prow) * cD + d0 + tx] = f2bf(tile[tx][prow]);
  }
}

// ---------------- bf16 WMMA GEMM:  C[M,N] = A[M,K] * W[N,K]^T + bias ----------------
// Block = 8 waves sharing one 64-wide N tile, spanning 128 rows of M.
// Per k-step the 64x32 bf16 B tile (4KB) is DMA'd to LDS with
// global_load_async_to_lds_b128 (one 16B chunk per thread), double-buffered,
// synchronized with s_wait_asynccnt + barrier, so the async copy of step k+1
// overlaps the 4 WMMAs of step k. B columns use an 80B stride in LDS: 16B
// alignment kept for ds_load_b128 and the 16 lanes of each half hit 16
// disjoint 4-bank groups (conflict-free). All four B fragments are read into
// distinct registers first (one ds clause + one wait), then the 4 WMMAs issue
// back-to-back.
__global__ __launch_bounds__(256) void k_gemm_bf16(const unsigned short* __restrict__ A,
                                                   const unsigned short* __restrict__ W,
                                                   const float* __restrict__ bias,
                                                   float* __restrict__ Cf,
                                                   unsigned short* __restrict__ Cbf,
                                                   int M, int N, int K, int relu) {
  __shared__ __attribute__((aligned(16))) unsigned short bsm[2][64 * 40]; // 2 x 5120B

  const int tid  = threadIdx.x;
  const int lane = tid & 31;
  const int wave = tid >> 5;
  const int tiles_n = N >> 6;
  const int tn = blockIdx.x % tiles_n;
  const int tm = (blockIdx.x / tiles_n) * 8 + wave;

  const int hs  = lane >> 4;     // half select (0/1)
  const int l16 = lane & 15;

  // async-copy mapping: 256 threads x 16B = 4KB = one 64x32 bf16 B tile
  const int brow = tid >> 2;          // 0..63 : column of B == row of W
  const int bchk = tid & 3;           // 0..3  : 16B chunk within the 64B row
  const unsigned lds_dst0 = (unsigned)(size_t)(&bsm[0][0]) + brow * 80 + bchk * 16;
  const unsigned lds_dst1 = (unsigned)(size_t)(&bsm[1][0]) + brow * 80 + bchk * 16;
  const unsigned long long wbase64 = (unsigned long long)(size_t)W;
  const unsigned wvoff = (unsigned)((long)(tn * 64 + brow) * K * 2 + bchk * 16);

  // A fragment source: lanes 0-15 row M=l16 with K halves {0-7,16-23},
  //                    lanes 16-31 same rows with K halves {8-15,24-31}
  const unsigned short* arow = A + (long)(tm * 16 + l16) * K + hs * 8;

  const int nk = K >> 5;

  // prologue: stage k-step 0 into buffer 0 (ASYNCcnt-tracked DMA to LDS)
  asm volatile("global_load_async_to_lds_b128 %0, %1, %2"
               :: "v"(lds_dst0), "v"(wvoff), "s"(wbase64) : "memory");

  v8f acc[4];
#pragma unroll
  for (int j = 0; j < 4; ++j) acc[j] = (v8f){0.f,0.f,0.f,0.f,0.f,0.f,0.f,0.f};

  for (int i = 0; i < nk; ++i) {
    // my DMA for buf[i&1] is done; barrier => everyone's chunks are in LDS
    asm volatile("s_wait_asynccnt 0x0" ::: "memory");
    __syncthreads();
    if (i + 1 < nk) {   // kick DMA for next k-step into the other buffer
      const unsigned voff = wvoff + (unsigned)((i + 1) * 64);
      const unsigned dst  = ((i + 1) & 1) ? lds_dst1 : lds_dst0;
      asm volatile("global_load_async_to_lds_b128 %0, %1, %2"
                   :: "v"(dst), "v"(voff), "s"(wbase64) : "memory");
    }
    const v8bf* pa = (const v8bf*)(arow + i * 32);
    v16bf afrag = mk16(pa[0], pa[2]);        // K +0..7 and +16..23 within lane's half
    const unsigned short* bb = bsm[i & 1];

    // read all 4 B fragments first (distinct registers -> one ds clause, one wait)
    v16bf bfrag[4];
#pragma unroll
    for (int j = 0; j < 4; ++j) {
      const v8bf* pw = (const v8bf*)(bb + (j * 16 + l16) * 40 + hs * 8);
      bfrag[j] = mk16(pw[0], pw[2]);         // K +0..7 and +16..23 within lane's half
    }
    // then 4 WMMAs back-to-back
#pragma unroll
    for (int j = 0; j < 4; ++j) {
      acc[j] = __builtin_amdgcn_wmma_f32_16x16x32_bf16(
          false, afrag, false, bfrag[j], (short)0, acc[j], false, false);
    }
  }

  // C layout: VGPR r -> row (tm*16 + r + hs*8), col (tn*64 + j*16 + l16)
  const int rbase = tm * 16 + hs * 8;
#pragma unroll
  for (int j = 0; j < 4; ++j) {
    const int col = tn * 64 + j * 16 + l16;
    const float bv = bias ? bias[col] : 0.f;
#pragma unroll
    for (int r = 0; r < 8; ++r) {
      float val = acc[j][r] + bv;
      if (relu) val = fmaxf(val, 0.f);
      const long idx = (long)(rbase + r) * N + col;
      if (Cf)  Cf[idx]  = val;
      if (Cbf) Cbf[idx] = f2bf(val);
    }
  }
}

// ---------------- MS-deformable sampling (one wave per (b,q,h), lanes = DH) ----------------
__global__ __launch_bounds__(256) void k_msda_sample(const float* __restrict__ v,    // (B,S,256)
                                                     const float* __restrict__ offs, // (B,NQ,256)
                                                     const float* __restrict__ aw,   // (B,NQ,128)
                                                     const float* __restrict__ qref, // (B,NQ,2)
                                                     unsigned short* __restrict__ out) { // (B,NQ,256) bf16
  const int wid  = blockIdx.x * 8 + (threadIdx.x >> 5);
  const int lane = threadIdx.x & 31;
  const int b    = wid >> 15;                // NQ*NH = 32768
  const int rem  = wid & 32767;
  const int q    = rem >> 3;
  const int h    = rem & 7;

  // softmax over 16 attention weights (wave-uniform; each lane redundantly)
  const float* awp = aw + (long)(b * cNQ + q) * 128 + h * 16;
  float w[16], mx = -1e30f;
#pragma unroll
  for (int i = 0; i < 16; ++i) { w[i] = awp[i]; mx = fmaxf(mx, w[i]); }
  float ssum = 0.f;
#pragma unroll
  for (int i = 0; i < 16; ++i) { w[i] = __expf(w[i] - mx); ssum += w[i]; }
  const float inv = 1.f / ssum;

  const float rx = qref[((long)b * cNQ + q) * 2 + 0];
  const float ry = qref[((long)b * cNQ + q) * 2 + 1];
  const float* offp = offs + (long)(b * cNQ + q) * 256 + h * 32;  // (l,p,2)

  const int dims[4] = {128, 64, 32, 16};
  const int lofs[4] = {0, 16384, 20480, 21504};

  float accv = 0.f;
#pragma unroll
  for (int l = 0; l < 4; ++l) {
    const int Wl = dims[l], Hl = dims[l];
    const float* vl = v + ((long)b * cS + lofs[l]) * cD + h * cDH + lane;
#pragma unroll
    for (int p = 0; p < 4; ++p) {
      const float ox = offp[(l * 4 + p) * 2 + 0];
      const float oy = offp[(l * 4 + p) * 2 + 1];
      const float x  = (rx + ox / (float)Wl) * (float)Wl - 0.5f;
      const float y  = (ry + oy / (float)Hl) * (float)Hl - 0.5f;
      const float x0 = floorf(x), y0 = floorf(y);
      const float wx = x - x0,    wy = y - y0;
      float sample = 0.f;
#pragma unroll
      for (int c = 0; c < 4; ++c) {
        const int dx = c & 1, dy = c >> 1;
        const float xc = x0 + dx, yc = y0 + dy;
        const float cw = (dx ? wx : 1.f - wx) * (dy ? wy : 1.f - wy);
        const bool valid = (xc >= 0.f) && (xc < (float)Wl) && (yc >= 0.f) && (yc < (float)Hl);
        if (valid && cw != 0.f) {            // wave-uniform branch; coalesced 128B row read
          const int xi = (int)xc, yi = (int)yc;
          sample += cw * vl[(long)(yi * Wl + xi) * cD];
        }
      }
      accv += w[l * 4 + p] * inv * sample;
    }
  }
  out[(long)(b * cNQ + q) * cD + h * cDH + lane] = f2bf(accv);
}

// ---------------- fused residual + LayerNorm (wave per row of 256) ----------------
__global__ __launch_bounds__(256) void k_add_ln(float* __restrict__ x,            // in/out qf
                                                const float* __restrict__ res,
                                                const float* __restrict__ g,
                                                const float* __restrict__ bta,
                                                unsigned short* __restrict__ out_bf) {
  const int row  = blockIdx.x * 8 + (threadIdx.x >> 5);
  const int lane = threadIdx.x & 31;
  const long base = (long)row * cD;
  float t[8], s = 0.f;
#pragma unroll
  for (int j = 0; j < 8; ++j) {
    const int c = lane + j * 32;
    t[j] = x[base + c] + res[base + c];
    s += t[j];
  }
#pragma unroll
  for (int o = 16; o > 0; o >>= 1) s += __shfl_xor(s, o, 32);
  const float mean = s * (1.f / 256.f);
  float vs = 0.f;
#pragma unroll
  for (int j = 0; j < 8; ++j) { const float d = t[j] - mean; vs += d * d; }
#pragma unroll
  for (int o = 16; o > 0; o >>= 1) vs += __shfl_xor(vs, o, 32);
  const float rstd = rsqrtf(vs * (1.f / 256.f) + 1e-5f);
#pragma unroll
  for (int j = 0; j < 8; ++j) {
    const int c = lane + j * 32;
    const float yv = (t[j] - mean) * rstd * g[c] + bta[c];
    x[base + c] = yv;
    if (out_bf) out_bf[base + c] = f2bf(yv);
  }
}

// ---------------- host orchestration ----------------
extern "C" void kernel_launch(void* const* d_in, const int* in_sizes, int n_in,
                              void* d_out, int out_size, void* d_ws, size_t ws_size,
                              hipStream_t stream) {
  // input dict order
  const float* src0   = (const float*)d_in[0];
  const float* src1   = (const float*)d_in[3];
  const float* src2i  = (const float*)d_in[6];
  const float* src3   = (const float*)d_in[9];
  const float* q_feat = (const float*)d_in[12];
  const float* q_pos  = (const float*)d_in[13];
  const float* q_ref  = (const float*)d_in[14];
  const float* so_w   = (const float*)d_in[16];
  const float* so_b   = (const float*)d_in[17];
  const float* aw_w   = (const float*)d_in[18];
  const float* aw_b   = (const float*)d_in[19];
  const float* vp_w   = (const float*)d_in[20];
  const float* vp_b   = (const float*)d_in[21];
  const float* op_w   = (const float*)d_in[22];
  const float* op_b   = (const float*)d_in[23];
  const float* n1_g   = (const float*)d_in[24];
  const float* n1_b   = (const float*)d_in[25];
  const float* l1_w   = (const float*)d_in[26];
  const float* l1_b   = (const float*)d_in[27];
  const float* l2_w   = (const float*)d_in[28];
  const float* l2_b   = (const float*)d_in[29];
  const float* n2_g   = (const float*)d_in[30];
  const float* n2_b   = (const float*)d_in[31];
  // masks/pos{l}/level_embed unused: masks are all-false -> valid_ratios == 1.

  // element counts
  const long E_VP = (long)cNL * cD * cD;
  const long E_SO = (long)cNL * cD * cD;
  const long E_AW = (long)cNL * 128 * cD;
  const long E_OP = (long)cNL * cD * cD;
  const long E_L1 = (long)cNL * cDFF * cD;
  const long E_L2 = (long)cNL * cD * cDFF;
  const long E_SF = (long)cB * cS * cD;      // src_flat / v
  const long E_Q  = (long)cB * cNQ * cD;     // 2,097,152
  const long E_AWB= (long)cB * cNQ * 128;
  const long E_H  = (long)cB * cNQ * cDFF;

  // workspace carve-up (256B aligned)
  size_t off = 0;
  auto carve = [&](size_t bytes) -> size_t {
    size_t o = off;
    off = (off + bytes + 255) & ~(size_t)255;
    return o;
  };
  char* ws = (char*)d_ws;
  unsigned short* wvp  = (unsigned short*)(ws + carve(E_VP * 2));
  unsigned short* wso  = (unsigned short*)(ws + carve(E_SO * 2));
  unsigned short* waw  = (unsigned short*)(ws + carve(E_AW * 2));
  unsigned short* wop  = (unsigned short*)(ws + carve(E_OP * 2));
  unsigned short* wl1  = (unsigned short*)(ws + carve(E_L1 * 2));
  unsigned short* wl2  = (unsigned short*)(ws + carve(E_L2 * 2));
  unsigned short* sflat= (unsigned short*)(ws + carve(E_SF * 2));
  float*          vbuf = (float*)         (ws + carve(E_SF * 4));
  unsigned short* qb   = (unsigned short*)(ws + carve(E_Q * 2));
  float*          offs = (float*)         (ws + carve(E_Q * 4));
  float*          awb  = (float*)         (ws + carve(E_AWB * 4));
  unsigned short* samp = (unsigned short*)(ws + carve(E_Q * 2));
  float*          resb = (float*)         (ws + carve(E_Q * 4));   // src2 / ff
  unsigned short* qfbf = (unsigned short*)(ws + carve(E_Q * 2));
  unsigned short* hbuf = (unsigned short*)(ws + carve(E_H * 2));
  (void)ws_size; (void)in_sizes; (void)n_in; (void)out_size;

  float* qf = (float*)d_out;   // qf lives in d_out; final LN leaves answer in place
  hipMemcpyAsync(qf, q_feat, (size_t)E_Q * 4, hipMemcpyDeviceToDevice, stream);

  // one-time weight conversion
  auto conv = [&](const float* s, unsigned short* d, long n) {
    k_convert_bf16<<<(int)((n + 255) / 256), 256, 0, stream>>>(s, d, n);
  };
  conv(vp_w, wvp, E_VP); conv(so_w, wso, E_SO); conv(aw_w, waw, E_AW);
  conv(op_w, wop, E_OP); conv(l1_w, wl1, E_L1); conv(l2_w, wl2, E_L2);

  // flatten value features once
  k_flatten<<<dim3(cS / 32, cD / 32, cB), dim3(32, 8), 0, stream>>>(src0, src1, src2i, src3, sflat);

  const int Mq = cB * cNQ;     // 8192
  const int Mv = cB * cS;      // 43520
  auto gemm = [&](const unsigned short* A, const unsigned short* Wt, const float* bias,
                  float* Cf, unsigned short* Cbf, int M, int N, int K, int relu) {
    const int blocks = (M / 128) * (N / 64);   // all M here are multiples of 128
    k_gemm_bf16<<<blocks, 256, 0, stream>>>(A, Wt, bias, Cf, Cbf, M, N, K, relu);
  };

  for (int l = 0; l < cNL; ++l) {
    // value projection: v = src_flat @ vp_w^T + vp_b     (43520 x 256 x 256)
    gemm(sflat, wvp + (long)l * cD * cD, vp_b + l * cD, vbuf, nullptr, Mv, cD, cD, 0);
    // q = bf16(qf + q_pos)
    k_add_bf16<<<(int)((E_Q + 255) / 256), 256, 0, stream>>>(qf, q_pos, qb, E_Q);
    // sampling offsets + attention logits
    gemm(qb, wso + (long)l * cD * cD, so_b + l * cD, offs, nullptr, Mq, cD, cD, 0);
    gemm(qb, waw + (long)l * 128 * cD, aw_b + l * 128, awb, nullptr, Mq, 128, cD, 0);
    // deformable bilinear sampling -> samp (bf16)
    k_msda_sample<<<(cB * cNQ * cNH) / 8, 256, 0, stream>>>(vbuf, offs, awb, q_ref, samp);
    // output projection
    gemm(samp, wop + (long)l * cD * cD, op_b + l * cD, resb, nullptr, Mq, cD, cD, 0);
    // qf = LN(qf + src2); also bf16 copy for FFN
    k_add_ln<<<Mq / 8, 256, 0, stream>>>(qf, resb, n1_g + l * cD, n1_b + l * cD, qfbf);
    // FFN: h = relu(qf @ l1_w^T + l1_b) (bf16), ff = h @ l2_w^T + l2_b
    gemm(qfbf, wl1 + (long)l * cDFF * cD, l1_b + l * cDFF, nullptr, hbuf, Mq, cDFF, cD, 1);
    gemm(hbuf, wl2 + (long)l * cD * cDFF, l2_b + l * cD, resb, nullptr, Mq, cD, cDFF, 0);
    // qf = LN(qf + ff)
    k_add_ln<<<Mq / 8, 256, 0, stream>>>(qf, resb, n2_g + l * cD, n2_b + l * cD, nullptr);
  }
}
